// Decoder_5763846111746
// MI455X (gfx1250) — matrix operand order repaired and checked by reading, not compiled, not run on hardware
//
#include <hip/hip_runtime.h>
#include <cstdint>
#include <cmath>

#define NEG_INF_F (-10000.0f)

typedef __attribute__((ext_vector_type(16))) __bf16 bf16x16;
typedef __attribute__((ext_vector_type(8)))  float  f32x8;

union FragAB {
  bf16x16 v;
  uint4   q[2];
};

__device__ __forceinline__ uint16_t f2bf(float f) {
  uint32_t u = __float_as_uint(f);
  uint32_t r = u + 0x7FFFu + ((u >> 16) & 1u);   // round-to-nearest-even
  return (uint16_t)(r >> 16);
}

__device__ __forceinline__ float sigmoidf_(float x) { return 1.0f / (1.0f + __expf(-x)); }

// ---------------------------------------------------------------------------
// Register-blocked GEMM: out[32, N] = A[32, K]_bf16 * W[N, K]_bf16^T (+ bias)
// One wave computes a 32x32 output tile (2 M-subtiles x 2 N-subtiles), so each
// A and B fragment is reused twice: 2 b128 loads per v_wmma_f32_16x16x32_bf16,
// and the W (embedding) stream out of L2 is halved vs 1-tile-per-wave.
// Requires N % 32 == 0 (both 2304 and Vpad=30528 satisfy this).
// Fragment layouts per CDNA5 ISA 7.12.2 (16-bit A 16x32, B 32x16, f32 C/D).
// ---------------------------------------------------------------------------
__global__ __launch_bounds__(128) void wmma_gemm_bf16(
    const uint16_t* __restrict__ A,    // [32, K] bf16 (row-major)
    const uint16_t* __restrict__ W,    // [N, K]  bf16 (row-major); acts as B^T
    const float*    __restrict__ bias, // [N] or nullptr
    float*          __restrict__ out,  // [32, N] fp32, ld = N
    int K, int N)
{
  const int lane = threadIdx.x & 31;
  const int wave = threadIdx.x >> 5;
  const int row  = lane & 15;   // M row (A) / N col (B,D)
  const int hi   = lane >> 4;   // K-half selector

  const int n0 = (blockIdx.x * 4 + wave) * 32;
  if (n0 >= N) return;                      // wave-uniform: EXEC stays all-ones

  const uint16_t* Arow0 = A + (size_t)row * K;          // batch rows 0..15
  const uint16_t* Arow1 = A + (size_t)(16 + row) * K;   // batch rows 16..31
  const uint16_t* Wrow0 = W + (size_t)(n0 + row) * K;        // cols n0..n0+15
  const uint16_t* Wrow1 = W + (size_t)(n0 + 16 + row) * K;   // cols n0+16..n0+31

  f32x8 c00 = {0.f,0.f,0.f,0.f,0.f,0.f,0.f,0.f};
  f32x8 c01 = c00, c10 = c00, c11 = c00;

  for (int k = 0; k < K; k += 32) {
    FragAB a0, a1, b0, b1;
    // A 16x32: lanes<16 hold K 0..7 / 16..23; lanes>=16 hold K 8..15 / 24..31
    a0.q[0] = *(const uint4*)(Arow0 + k + hi * 8);
    a0.q[1] = *(const uint4*)(Arow0 + k + 16 + hi * 8);
    a1.q[0] = *(const uint4*)(Arow1 + k + hi * 8);
    a1.q[1] = *(const uint4*)(Arow1 + k + 16 + hi * 8);
    // B 32x16: lanes<16 hold K 0..7 / 16..23 of the lane's column; lanes>=16 K 8..15 / 24..31
    const uint16_t* wp0 = Wrow0 + k + hi * 8;
    const uint16_t* wp1 = Wrow1 + k + hi * 8;
    b0.q[0] = ((const uint4*)wp0)[0];
    b0.q[1] = ((const uint4*)(wp0 + 16))[0];
    b1.q[0] = ((const uint4*)wp1)[0];
    b1.q[1] = ((const uint4*)(wp1 + 16))[0];

    c00 = __builtin_amdgcn_wmma_f32_16x16x32_bf16(false, a0.v, false, b0.v,
                                                  (short)0, c00, false, false);
    c01 = __builtin_amdgcn_wmma_f32_16x16x32_bf16(false, a0.v, false, b1.v,
                                                  (short)0, c01, false, false);
    c10 = __builtin_amdgcn_wmma_f32_16x16x32_bf16(false, a1.v, false, b0.v,
                                                  (short)0, c10, false, false);
    c11 = __builtin_amdgcn_wmma_f32_16x16x32_bf16(false, a1.v, false, b1.v,
                                                  (short)0, c11, false, false);
  }

  // D 16x16 f32: vgpr r -> M = r (lanes 0-15) / M = 8+r (lanes 16-31); N = lane%16
  const int ncol0 = n0 + row;
  const int ncol1 = n0 + 16 + row;
  const float bv0 = bias ? bias[ncol0] : 0.0f;
  const float bv1 = bias ? bias[ncol1] : 0.0f;
  float* o0 = out + (size_t)(hi * 8) * N;        // M-subtile 0 base
  float* o1 = out + (size_t)(16 + hi * 8) * N;   // M-subtile 1 base
#pragma unroll
  for (int r = 0; r < 8; ++r) {
    o0[(size_t)r * N + ncol0] = c00[r] + bv0;
    o0[(size_t)r * N + ncol1] = c01[r] + bv1;
    o1[(size_t)r * N + ncol0] = c10[r] + bv0;
    o1[(size_t)r * N + ncol1] = c11[r] + bv1;
  }
}

// ---------------------------------------------------------------------------
// Elementwise / helper kernels
// ---------------------------------------------------------------------------
__global__ void k_f32_to_bf16_pad_rows(const float* __restrict__ src,
                                       uint16_t* __restrict__ dst,
                                       int rowsValid, int rowsPad, int cols) {
  long long idx = (long long)blockIdx.x * blockDim.x + threadIdx.x;
  long long total = (long long)rowsPad * cols;
  if (idx >= total) return;
  int r = (int)(idx / cols);
  int c = (int)(idx % cols);
  dst[idx] = (r < rowsValid) ? f2bf(src[(size_t)r * cols + c]) : (uint16_t)0;
}

__global__ void k_init_h(const float* __restrict__ hid, float* __restrict__ h,
                         uint16_t* __restrict__ hb, int n) {
  int i = blockIdx.x * blockDim.x + threadIdx.x;
  if (i < n) { float f = hid[i]; h[i] = f; hb[i] = f2bf(f); }
}

__global__ void k_set_w(const float* __restrict__ dec, int u, int NU, int H,
                        float* __restrict__ w, uint16_t* __restrict__ wb, int n) {
  int i = blockIdx.x * blockDim.x + threadIdx.x;
  if (i >= n) return;
  int b = i / H, j = i % H;
  float f = dec[((size_t)b * NU + u) * H + j];
  w[i] = f; wb[i] = f2bf(f);
}

__global__ void k_gru_combine(const float* __restrict__ gi, const float* __restrict__ gh,
                              float* __restrict__ h, uint16_t* __restrict__ hb,
                              int H, int n) {
  int i = blockIdx.x * blockDim.x + threadIdx.x;
  if (i >= n) return;
  int b = i / H, j = i % H;
  const float* gib = gi + (size_t)b * 3 * H;
  const float* ghb = gh + (size_t)b * 3 * H;
  float r  = sigmoidf_(gib[j] + ghb[j]);
  float z  = sigmoidf_(gib[H + j] + ghb[H + j]);
  float nn = tanhf(gib[2 * H + j] + r * ghb[2 * H + j]);
  float hv = (1.0f - z) * nn + z * h[i];
  h[i] = hv; hb[i] = f2bf(hv);
}

// attention scores + masked softmax + context + p_gen, one block per batch row
__global__ __launch_bounds__(256) void k_attn(
    const float* __restrict__ enc, const int* __restrict__ ids,
    const float* __restrict__ h, const float* __restrict__ w,
    const float* __restrict__ wgw, const float* __restrict__ wgb,
    float* __restrict__ attn_hist, float* __restrict__ p_gen, int S, int H)
{
  extern __shared__ float smem[];
  float* h_sh   = smem;             // [H]
  float* ctx_sh = smem + H;         // [H]
  float* a_sh   = smem + 2 * H;     // [S]
  float* red    = smem + 2 * H + S; // [256]
  const int b = blockIdx.x;
  const int tid = threadIdx.x;

  for (int j = tid; j < H; j += blockDim.x) h_sh[j] = h[(size_t)b * H + j];
  __syncthreads();

  for (int s = tid; s < S; s += blockDim.x) {
    const float* er = enc + ((size_t)b * S + s) * H;
    float e = 0.f;
    for (int j = 0; j < H; ++j) e += er[j] * h_sh[j];
    if (ids[(size_t)b * S + s] == 0) e = NEG_INF_F;
    a_sh[s] = e;
  }
  __syncthreads();

  float m = -3.4e38f;
  for (int s = tid; s < S; s += blockDim.x) m = fmaxf(m, a_sh[s]);
  red[tid] = m; __syncthreads();
  for (int st = blockDim.x >> 1; st > 0; st >>= 1) {
    if (tid < st) red[tid] = fmaxf(red[tid], red[tid + st]);
    __syncthreads();
  }
  m = red[0]; __syncthreads();

  float ssum = 0.f;
  for (int s = tid; s < S; s += blockDim.x) { float e = __expf(a_sh[s] - m); a_sh[s] = e; ssum += e; }
  red[tid] = ssum; __syncthreads();
  for (int st = blockDim.x >> 1; st > 0; st >>= 1) {
    if (tid < st) red[tid] += red[tid + st];
    __syncthreads();
  }
  float inv = 1.0f / red[0]; __syncthreads();

  for (int s = tid; s < S; s += blockDim.x) {
    float a = a_sh[s] * inv; a_sh[s] = a; attn_hist[(size_t)b * S + s] = a;
  }
  __syncthreads();

  for (int j = tid; j < H; j += blockDim.x) {
    float acc = 0.f;
    const float* ej = enc + (size_t)b * S * H + j;
    for (int s = 0; s < S; ++s) acc += a_sh[s] * ej[(size_t)s * H];
    ctx_sh[j] = acc;
  }
  __syncthreads();

  float acc = 0.f;
  for (int i = tid; i < 3 * H; i += blockDim.x) {
    float x = (i < H) ? w[(size_t)b * H + i]
                      : (i < 2 * H ? h_sh[i - H] : ctx_sh[i - 2 * H]);
    acc += x * wgw[i];
  }
  red[tid] = acc; __syncthreads();
  for (int st = blockDim.x >> 1; st > 0; st >>= 1) {
    if (tid < st) red[tid] += red[tid + st];
    __syncthreads();
  }
  if (tid == 0) p_gen[b] = sigmoidf_(red[0] + wgb[0]);
}

__global__ __launch_bounds__(256) void k_rowstats(const float* __restrict__ logits,
                                                  int V, int Vpad,
                                                  float* __restrict__ rmax,
                                                  float* __restrict__ rinv) {
  __shared__ float red[256];
  int b = blockIdx.x, tid = threadIdx.x;
  const float* lr = logits + (size_t)b * Vpad;
  float m = -3.4e38f;
  for (int v = tid; v < V; v += 256) m = fmaxf(m, lr[v]);
  red[tid] = m; __syncthreads();
  for (int st = 128; st > 0; st >>= 1) {
    if (tid < st) red[tid] = fmaxf(red[tid], red[tid + st]);
    __syncthreads();
  }
  m = red[0]; __syncthreads();
  float s = 0.f;
  for (int v = tid; v < V; v += 256) s += __expf(lr[v] - m);
  red[tid] = s; __syncthreads();
  for (int st = 128; st > 0; st >>= 1) {
    if (tid < st) red[tid] += red[tid + st];
    __syncthreads();
  }
  if (tid == 0) { rmax[b] = m; rinv[b] = 1.0f / red[0]; }
}

__global__ void k_pfinal(const float* __restrict__ logits, const float* __restrict__ rmax,
                         const float* __restrict__ rinv, const float* __restrict__ pg,
                         float* __restrict__ out, int V, int Vpad, int NU, int MV,
                         int u, int t, long long total) {
  long long i = (long long)blockIdx.x * blockDim.x + threadIdx.x;
  if (i >= total) return;
  int b = (int)(i / V), v = (int)(i % V);
  float p = pg[b] * __expf(logits[(size_t)b * Vpad + v] - rmax[b]) * rinv[b];
  out[(((size_t)b * NU + u) * MV + t) * (size_t)V + v] = p;
}

__global__ void k_scatter(const int* __restrict__ ids, const float* __restrict__ ah,
                          const float* __restrict__ pg, float* __restrict__ out,
                          int S, int V, int NU, int MV, int u, int t, int total) {
  int i = blockIdx.x * blockDim.x + threadIdx.x;
  if (i >= total) return;
  int b = i / S;
  float val = (1.0f - pg[b]) * ah[i];
  size_t o = (((size_t)b * NU + u) * MV + t) * (size_t)V + ids[i];
  atomicAdd(&out[o], val);
}

__global__ __launch_bounds__(256) void k_argmax(const float* __restrict__ out,
                                                int V, int NU, int MV, int u, int t,
                                                int* __restrict__ widx) {
  __shared__ float rv[256];
  __shared__ int   ri[256];
  int b = blockIdx.x, tid = threadIdx.x;
  const float* row = out + (((size_t)b * NU + u) * MV + t) * (size_t)V;
  float bv = -3.4e38f; int bi = 0x7fffffff;
  for (int v = tid; v < V; v += 256) {
    float f = row[v];
    if (f > bv || (f == bv && v < bi)) { bv = f; bi = v; }
  }
  rv[tid] = bv; ri[tid] = bi; __syncthreads();
  for (int st = 128; st > 0; st >>= 1) {
    if (tid < st) {
      if (rv[tid + st] > rv[tid] ||
          (rv[tid + st] == rv[tid] && ri[tid + st] < ri[tid])) {
        rv[tid] = rv[tid + st]; ri[tid] = ri[tid + st];
      }
    }
    __syncthreads();
  }
  if (tid == 0) widx[b] = ri[0];
}

__global__ void k_gather_embed(const float* __restrict__ E, const int* __restrict__ widx,
                               float* __restrict__ w, uint16_t* __restrict__ wb,
                               int H, int n) {
  int i = blockIdx.x * blockDim.x + threadIdx.x;
  if (i >= n) return;
  int b = i / H, j = i % H;
  float f = E[(size_t)widx[b] * H + j];
  w[i] = f; wb[i] = f2bf(f);
}

// ---------------------------------------------------------------------------
extern "C" void kernel_launch(void* const* d_in, const int* in_sizes, int n_in,
                              void* d_out, int out_size, void* d_ws, size_t ws_size,
                              hipStream_t stream) {
  const int*   input_ids = (const int*)d_in[0];
  const float* dec  = (const float*)d_in[1];
  const float* enc  = (const float*)d_in[2];
  const float* hid  = (const float*)d_in[3];
  const float* E    = (const float*)d_in[4];
  const float* wih  = (const float*)d_in[5];
  const float* whh  = (const float*)d_in[6];
  const float* bih  = (const float*)d_in[7];
  const float* bhh  = (const float*)d_in[8];
  const float* wgw  = (const float*)d_in[9];
  const float* wgb  = (const float*)d_in[10];

  const int H  = (int)lround(sqrt((double)in_sizes[5] / 3.0));   // 768
  const int B  = in_sizes[3] / H;                                // 32
  const int S  = in_sizes[0] / B;                                // 256
  const int V  = in_sizes[4] / H;                                // 30522
  const int NU = in_sizes[1] / (B * H);                          // 5
  const int MV = out_size / (B * NU * V);                        // 4
  const int H3 = 3 * H;
  const int Vpad = (V + 31) & ~31;                               // 30528

  char* ws = (char*)d_ws;
  size_t off = 0;
  auto alloc = [&](size_t bytes) {
    char* p = ws + off;
    off = (off + bytes + 255) & ~(size_t)255;
    return p;
  };
  uint16_t* Eb     = (uint16_t*)alloc((size_t)Vpad * H * 2);
  uint16_t* Wihb   = (uint16_t*)alloc((size_t)H3 * H * 2);
  uint16_t* Whhb   = (uint16_t*)alloc((size_t)H3 * H * 2);
  float*    h      = (float*)   alloc((size_t)B * H * 4);
  uint16_t* hb     = (uint16_t*)alloc((size_t)B * H * 2);
  float*    w      = (float*)   alloc((size_t)B * H * 4);
  uint16_t* wb     = (uint16_t*)alloc((size_t)B * H * 2);
  float*    gi     = (float*)   alloc((size_t)B * H3 * 4);
  float*    gh     = (float*)   alloc((size_t)B * H3 * 4);
  float*    logits = (float*)   alloc((size_t)B * Vpad * 4);
  float*    ah     = (float*)   alloc((size_t)B * S * 4);
  float*    pg     = (float*)   alloc((size_t)B * 4);
  float*    rmax   = (float*)   alloc((size_t)B * 4);
  float*    rinv   = (float*)   alloc((size_t)B * 4);
  int*      widx   = (int*)     alloc((size_t)B * 4);
  (void)ws_size; (void)n_in;

  float* out = (float*)d_out;

  // One-time weight conversions (bf16; embed L2-resident at 46.9 MB)
  {
    long long tot = (long long)Vpad * H;
    k_f32_to_bf16_pad_rows<<<(int)((tot + 255) / 256), 256, 0, stream>>>(E, Eb, V, Vpad, H);
  }
  {
    long long tot = (long long)H3 * H;
    k_f32_to_bf16_pad_rows<<<(int)((tot + 255) / 256), 256, 0, stream>>>(wih, Wihb, H3, H3, H);
    k_f32_to_bf16_pad_rows<<<(int)((tot + 255) / 256), 256, 0, stream>>>(whh, Whhb, H3, H3, H);
  }
  {
    int n = B * H;
    k_init_h<<<(n + 255) / 256, 256, 0, stream>>>(hid, h, hb, n);
  }

  const size_t attn_smem = (size_t)(2 * H + S + 256) * 4;  // 8 KB
  const int gemm_gru_blocks = (H3 / 32 + 3) / 4;           // 72 tiles -> 18 blocks
  const int gemm_voc_blocks = (Vpad / 32 + 3) / 4;         // 954 tiles -> 239 blocks

  for (int u = 0; u < NU; ++u) {
    {
      int n = B * H;
      k_set_w<<<(n + 255) / 256, 256, 0, stream>>>(dec, u, NU, H, w, wb, n);
    }
    for (int t = 0; t < MV; ++t) {
      // GRU gates via WMMA (32x32 tile per wave)
      wmma_gemm_bf16<<<gemm_gru_blocks, 128, 0, stream>>>(wb, Wihb, bih, gi, H, H3);
      wmma_gemm_bf16<<<gemm_gru_blocks, 128, 0, stream>>>(hb, Whhb, bhh, gh, H, H3);
      {
        int n = B * H;
        k_gru_combine<<<(n + 255) / 256, 256, 0, stream>>>(gi, gh, h, hb, H, n);
      }
      // attention softmax + context + p_gen
      k_attn<<<B, 256, attn_smem, stream>>>(enc, input_ids, h, w, wgw, wgb, ah, pg, S, H);
      // vocab logits via WMMA (dominant GEMM, bf16 embed streamed from L2,
      // each B fragment reused across both batch tiles -> E traffic halved)
      wmma_gemm_bf16<<<gemm_voc_blocks, 128, 0, stream>>>(hb, Eb, nullptr, logits, H, Vpad);
      k_rowstats<<<B, 256, 0, stream>>>(logits, V, Vpad, rmax, rinv);
      {
        long long tot = (long long)B * V;
        k_pfinal<<<(int)((tot + 255) / 256), 256, 0, stream>>>(
            logits, rmax, rinv, pg, out, V, Vpad, NU, MV, u, t, tot);
      }
      {
        int tot = B * S;
        k_scatter<<<(tot + 255) / 256, 256, 0, stream>>>(
            input_ids, ah, pg, out, S, V, NU, MV, u, t, tot);
      }
      k_argmax<<<B, 256, 0, stream>>>(out, V, NU, MV, u, t, widx);
      {
        int n = B * H;
        k_gather_embed<<<(n + 255) / 256, 256, 0, stream>>>(E, widx, w, wb, H, n);
      }
    }
  }
}
